// Attention_79259326480594
// MI455X (gfx1250) — compile-verified, matrix operand order checked by reading, not run on hardware
//
#include <hip/hip_runtime.h>
#include <hip/hip_bf16.h>

// ---------------------------------------------------------------------------
// Attention forward for MI455X (gfx1250, wave32, WMMA, async LDS staging).
//   Q = x@Wq^T + bq ; K = x@Wk^T + bk ; V = x@Wv^T + bv      (bf16 WMMA, f32 acc)
//   S = Q@K^T * 1/sqrt(64)   (f32 out)
//   P = softmax_rows(S)      (bf16 out)
//   O = P@V                  (f32 out -> d_out)
// ---------------------------------------------------------------------------

typedef __attribute__((ext_vector_type(16))) __bf16 v16bf;
typedef __attribute__((ext_vector_type(8)))  float  v8f;
typedef __attribute__((ext_vector_type(4)))  int    v4i;

#define D_MODEL 1024
#define BATCH   4
#define SEQ     2048

#define GAS __attribute__((address_space(1)))
#define LAS __attribute__((address_space(3)))

// --- async global->LDS copy of 8 bf16 (128 bits), ASYNCcnt-tracked --------
__device__ __forceinline__ void async_copy16(const __bf16* g, __bf16* l) {
#if __has_builtin(__builtin_amdgcn_global_load_async_to_lds_b128)
  __builtin_amdgcn_global_load_async_to_lds_b128(
      (GAS v4i*)g, (LAS v4i*)l, 0, 0);
#else
  *(uint4*)l = *(const uint4*)g;   // sync fallback
#endif
}

__device__ __forceinline__ void wait_async0() {
#if __has_builtin(__builtin_amdgcn_s_wait_asynccnt)
  __builtin_amdgcn_s_wait_asynccnt(0);
#else
  asm volatile("s_wait_asynccnt 0x0" ::: "memory");
#endif
}

// ---------------------------------------------------------------------------
// fp32 -> bf16 convert (vectorized x4, grid-stride)
// ---------------------------------------------------------------------------
struct bf16x4 { __bf16 v[4]; };

__global__ __launch_bounds__(256)
void cvt_f32_bf16(const float* __restrict__ in, __bf16* __restrict__ out, long n4) {
  long i = (long)blockIdx.x * blockDim.x + threadIdx.x;
  long stride = (long)gridDim.x * blockDim.x;
  for (; i < n4; i += stride) {
    float4 f = ((const float4*)in)[i];
    bf16x4 o;
    o.v[0] = (__bf16)f.x; o.v[1] = (__bf16)f.y;
    o.v[2] = (__bf16)f.z; o.v[3] = (__bf16)f.w;
    ((bf16x4*)out)[i] = o;
  }
}

// ---------------------------------------------------------------------------
// Tiled WMMA GEMM:  C[M,N] = scale * (A[M,K] x B^T) (+ bias[N])
//   TRANS_B = false : B stored [N,K] row-major (NT gemm: Linear / Q@K^T)
//   TRANS_B = true  : B stored [K,N] row-major (P@V), transposed into LDS
// BM=128, BN=128, BK=32; 256 threads = 8 waves; wave grid 4(M) x 2(N);
// each wave computes a 32x64 quadrant = 8x v_wmma_f32_16x16x32_bf16 per K-step.
// Double-buffered LDS, async global->LDS staging, one barrier per K-step.
// ---------------------------------------------------------------------------
template<bool TRANS_B, bool BIAS, bool OUT_BF16>
__global__ __launch_bounds__(256)
void gemm_wmma(const __bf16* __restrict__ Abase, long batchA,
               const __bf16* __restrict__ Bbase, long batchB,
               const float* __restrict__ bias,
               void* __restrict__ Cbase, long batchC,
               int M, int N, int K, int lda, int ldb, int ldc, float scale)
{
  constexpr int BM = 128, BN = 128, BK = 32;
  __shared__ __bf16 lds_a[2][BM * BK];
  __shared__ __bf16 lds_b[2][BN * BK];

  const __bf16* A = Abase + batchA * blockIdx.z;
  const __bf16* B = Bbase + batchB * blockIdx.z;

  const int bm   = blockIdx.y * BM;
  const int bn   = blockIdx.x * BN;
  const int tid  = threadIdx.x;
  const int lane = tid & 31;
  const int wid  = tid >> 5;
  const int wm   = wid >> 1;   // 0..3 (M)
  const int wn   = wid & 1;    // 0..1 (N)

  v8f acc[2][4] = {};

  // global->LDS staging coordinates
  const int arow = tid >> 1;          // 0..127
  const int acol = (tid & 1) * 16;    // 0 / 16
  const int tkk  = tid >> 3;          // 0..31   (TRANS_B path)
  const int tnn  = (tid & 7) * 16;    // 0..112

  auto stage = [&](int buf, int kb) {
    // A tile: 128x32, 32B per thread via two async b128
    const __bf16* ag = A + (long)(bm + arow) * lda + kb + acol;
    async_copy16(ag,     &lds_a[buf][arow * BK + acol]);
    async_copy16(ag + 8, &lds_a[buf][arow * BK + acol + 8]);
    if (!TRANS_B) {
      // B tile: 128x32 (K-major already)
      const __bf16* bg = B + (long)(bn + arow) * ldb + kb + acol;
      async_copy16(bg,     &lds_b[buf][arow * BK + acol]);
      async_copy16(bg + 8, &lds_b[buf][arow * BK + acol + 8]);
    } else {
      // B tile from [K=32, N=128]: load 16 N-contiguous, store transposed
      const __bf16* bg = B + (long)(kb + tkk) * ldb + bn + tnn;
      __bf16 tmp[16];
      *(uint4*)&tmp[0] = *(const uint4*)bg;
      *(uint4*)&tmp[8] = *(const uint4*)(bg + 8);
#pragma unroll
      for (int j = 0; j < 16; ++j)
        lds_b[buf][(tnn + j) * BK + tkk] = tmp[j];
    }
  };

  const int nk = K / BK;
  stage(0, 0);

  const int frow = lane & 15;
  const int fcol = (lane >> 4) * 16;

  for (int step = 0; step < nk; ++step) {
    const int cur = step & 1;
    wait_async0();
    __syncthreads();
    if (step + 1 < nk) stage(cur ^ 1, (step + 1) * BK);

    v16bf afrag[2], bfrag[4];
#pragma unroll
    for (int i = 0; i < 2; ++i)
      afrag[i] = *(const v16bf*)&lds_a[cur][(wm * 32 + i * 16 + frow) * BK + fcol];
#pragma unroll
    for (int j = 0; j < 4; ++j)
      bfrag[j] = *(const v16bf*)&lds_b[cur][(wn * 64 + j * 16 + frow) * BK + fcol];

#pragma unroll
    for (int i = 0; i < 2; ++i)
#pragma unroll
      for (int j = 0; j < 4; ++j)
        acc[i][j] = __builtin_amdgcn_wmma_f32_16x16x32_bf16(
            false, afrag[i], false, bfrag[j], (short)0, acc[i][j], false, false);
  }

  // ---- epilogue: C/D layout -> VGPR r: M = tile + (lane>=16 ? 8 : 0) + r, N = lane&15
  const int r8 = (lane >> 4) * 8;
  const int cn = lane & 15;
  const long cbase = batchC * blockIdx.z;
#pragma unroll
  for (int i = 0; i < 2; ++i) {
#pragma unroll
    for (int j = 0; j < 4; ++j) {
      const int col = bn + wn * 64 + j * 16 + cn;
      const float badd = BIAS ? bias[col] : 0.0f;
#pragma unroll
      for (int r = 0; r < 8; ++r) {
        const int row = bm + wm * 32 + i * 16 + r8 + r;
        const float v = acc[i][j][r] * scale + badd;
        if (OUT_BF16)
          ((__bf16*)Cbase)[cbase + (long)row * ldc + col] = (__bf16)v;
        else
          ((float*)Cbase)[cbase + (long)row * ldc + col] = v;
      }
    }
  }
}

// ---------------------------------------------------------------------------
// Row softmax over 2048 fp32 -> bf16. One 256-thread block per row.
// ---------------------------------------------------------------------------
__global__ __launch_bounds__(256)
void softmax_rows(const float* __restrict__ scores, __bf16* __restrict__ probs)
{
  const long row = blockIdx.x;
  const float* src = scores + row * (long)SEQ;
  __bf16*      dst = probs  + row * (long)SEQ;
  const int tid  = threadIdx.x;
  const int lane = tid & 31;
  const int wid  = tid >> 5;
  __shared__ float red[8];

  float v[8];
  float m = -3.0e38f;
#pragma unroll
  for (int i = 0; i < 8; ++i) {
    v[i] = src[tid + i * 256];
    m = fmaxf(m, v[i]);
  }
#pragma unroll
  for (int off = 16; off > 0; off >>= 1)
    m = fmaxf(m, __shfl_xor(m, off, 32));
  if (lane == 0) red[wid] = m;
  __syncthreads();
  if (wid == 0) {
    float t = (lane < 8) ? red[lane] : -3.0e38f;
#pragma unroll
    for (int off = 4; off > 0; off >>= 1)
      t = fmaxf(t, __shfl_xor(t, off, 32));
    if (lane == 0) red[0] = t;
  }
  __syncthreads();
  m = red[0];
  __syncthreads();

  float s = 0.0f;
#pragma unroll
  for (int i = 0; i < 8; ++i) { v[i] = __expf(v[i] - m); s += v[i]; }
#pragma unroll
  for (int off = 16; off > 0; off >>= 1)
    s += __shfl_xor(s, off, 32);
  if (lane == 0) red[wid] = s;
  __syncthreads();
  if (wid == 0) {
    float t = (lane < 8) ? red[lane] : 0.0f;
#pragma unroll
    for (int off = 4; off > 0; off >>= 1)
      t += __shfl_xor(t, off, 32);
    if (lane == 0) red[0] = t;
  }
  __syncthreads();
  const float inv = 1.0f / red[0];
#pragma unroll
  for (int i = 0; i < 8; ++i)
    dst[tid + i * 256] = (__bf16)(v[i] * inv);
}

// ---------------------------------------------------------------------------
extern "C" void kernel_launch(void* const* d_in, const int* in_sizes, int n_in,
                              void* d_out, int out_size, void* d_ws, size_t ws_size,
                              hipStream_t stream) {
  const float* x  = (const float*)d_in[0];
  const float* Wq = (const float*)d_in[1];
  const float* bq = (const float*)d_in[2];
  const float* Wk = (const float*)d_in[3];
  const float* bk = (const float*)d_in[4];
  const float* Wv = (const float*)d_in[5];
  const float* bv = (const float*)d_in[6];

  const long MS = (long)BATCH * SEQ;           // 8192 rows
  const long DD = (long)D_MODEL * D_MODEL;     // 1M weight elements

  char* ws = (char*)d_ws;
  __bf16* xb     = (__bf16*)ws;  ws += MS * D_MODEL * sizeof(__bf16);
  __bf16* Wqb    = (__bf16*)ws;  ws += DD * sizeof(__bf16);
  __bf16* Wkb    = (__bf16*)ws;  ws += DD * sizeof(__bf16);
  __bf16* Wvb    = (__bf16*)ws;  ws += DD * sizeof(__bf16);
  __bf16* Qb     = (__bf16*)ws;  ws += MS * D_MODEL * sizeof(__bf16);
  __bf16* Kb     = (__bf16*)ws;  ws += MS * D_MODEL * sizeof(__bf16);
  __bf16* Vb     = (__bf16*)ws;  ws += MS * D_MODEL * sizeof(__bf16);
  float*  scores = (float*)ws;   ws += (long)BATCH * SEQ * SEQ * sizeof(float);
  __bf16* probs  = (__bf16*)ws;

  // --- 1) downcast x and weights to bf16
  cvt_f32_bf16<<<4096, 256, 0, stream>>>(x,  xb,  MS * D_MODEL / 4);
  cvt_f32_bf16<<<1024, 256, 0, stream>>>(Wq, Wqb, DD / 4);
  cvt_f32_bf16<<<1024, 256, 0, stream>>>(Wk, Wkb, DD / 4);
  cvt_f32_bf16<<<1024, 256, 0, stream>>>(Wv, Wvb, DD / 4);

  // --- 2) QKV projections: [8192,1024] x [1024,1024]^T + bias -> bf16
  {
    dim3 grid(D_MODEL / 128, MS / 128, 1);
    gemm_wmma<false, true, true><<<grid, 256, 0, stream>>>(
        xb, 0, Wqb, 0, bq, Qb, 0, (int)MS, D_MODEL, D_MODEL,
        D_MODEL, D_MODEL, D_MODEL, 1.0f);
    gemm_wmma<false, true, true><<<grid, 256, 0, stream>>>(
        xb, 0, Wkb, 0, bk, Kb, 0, (int)MS, D_MODEL, D_MODEL,
        D_MODEL, D_MODEL, D_MODEL, 1.0f);
    gemm_wmma<false, true, true><<<grid, 256, 0, stream>>>(
        xb, 0, Wvb, 0, bv, Vb, 0, (int)MS, D_MODEL, D_MODEL,
        D_MODEL, D_MODEL, D_MODEL, 1.0f);
  }

  // --- 3) scores = Q @ K^T * 1/sqrt(64) -> f32   (per batch)
  {
    dim3 grid(SEQ / 128, SEQ / 128, BATCH);
    gemm_wmma<false, false, false><<<grid, 256, 0, stream>>>(
        Qb, (long)SEQ * D_MODEL, Kb, (long)SEQ * D_MODEL, nullptr,
        scores, (long)SEQ * SEQ, SEQ, SEQ, D_MODEL,
        D_MODEL, D_MODEL, SEQ, 0.125f);
  }

  // --- 4) row softmax -> bf16 probs
  softmax_rows<<<(int)MS, 256, 0, stream>>>(scores, probs);

  // --- 5) out = P @ V -> f32 d_out   (B operand is [K,N]: transpose on stage)
  {
    dim3 grid(D_MODEL / 128, SEQ / 128, BATCH);
    gemm_wmma<true, false, false><<<grid, 256, 0, stream>>>(
        probs, (long)SEQ * SEQ, Vb, (long)SEQ * D_MODEL, nullptr,
        d_out, (long)SEQ * D_MODEL, SEQ, D_MODEL, SEQ,
        SEQ, D_MODEL, D_MODEL, 1.0f);
  }
}
